// MultiHeadHypergraphAttention_29978871726114
// MI455X (gfx1250) — compile-verified, matrix operand order checked by reading, not run on hardware
//
#include <hip/hip_runtime.h>
#include <hip/hip_bf16.h>

// ---------------------------------------------------------------------------
// MI455X (gfx1250) fused multi-head hypergraph attention.
// All matmuls via v_wmma_f32_16x16x32_bf16 (wave32).
// ---------------------------------------------------------------------------

typedef __bf16        v16bf __attribute__((ext_vector_type(16)));
typedef float         v8f   __attribute__((ext_vector_type(8)));
typedef unsigned int  v8u   __attribute__((ext_vector_type(8)));

#define BS      4
#define N_EDGES 1024
#define N_NODES 4096
#define D_MODEL 512
#define N_HEADS 8
#define D_K     64

__device__ __forceinline__ unsigned short f2bf(float f) {
    // round-to-nearest-even f32 -> bf16
    unsigned int u = __float_as_uint(f);
    unsigned int r = u + 0x7FFFu + ((u >> 16) & 1u);
    return (unsigned short)(r >> 16);
}

// K index inside a 16-bit A/B WMMA fragment for VGPR v (pair base), half hh.
// ISA 7.12.2: VGPR0 K=0,1 (half0) / K=8,9 (half1); VGPR4..7 add +16.
__device__ __forceinline__ int wmma_kf(int v, int hh) {
    return ((v >> 2) << 4) | ((v & 3) << 1) | (hh << 3);
}

__device__ __forceinline__ v8f wmma_bf16(v16bf a, v16bf b, v8f c) {
    return __builtin_amdgcn_wmma_f32_16x16x32_bf16(
        false, a, false, b, (short)0, c, false, false);
}

// ---------------------------------------------------------------------------
// GEMM: C[M][N] = A[M][K] @ W[N][K]^T + bias   (torch Linear), N = K = 512.
// MODE 0: store f32 row-major.  MODE 1: store bf16 row-major.
// MODE 2: store bf16 transposed per batch of N_NODES rows:
//         out[b][col][row & 4095], b = row >> 12   (shift/mask, no idiv).
// Block: 256 thr (8 waves). Tile: 128(M) x 32(N), K-step 32.
// ---------------------------------------------------------------------------
#define ASTR 40   // padded LDS row stride (bf16 elems) to spread banks

template <int MODE>
__global__ __launch_bounds__(256) void gemm_xwt_bias_kernel(
    const float* __restrict__ A, const float* __restrict__ W,
    const float* __restrict__ bias,
    float* __restrict__ outF, unsigned short* __restrict__ outB)
{
    constexpr int K = D_MODEL;
    constexpr int N = D_MODEL;

    __shared__ unsigned short As[128 * ASTR];
    __shared__ unsigned short Ws[32 * ASTR];

    const int tid  = threadIdx.x;
    const int wv   = tid >> 5;
    const int lane = tid & 31;
    const int hh   = lane >> 4;
    const int ml   = lane & 15;
    const int bm   = blockIdx.x * 128;
    const int bn   = blockIdx.y * 32;

    v8f acc0 = {}, acc1 = {};

    for (int k0 = 0; k0 < K; k0 += 32) {
        __syncthreads();   // protect LDS reuse from previous iteration
        // stage A tile: 128x32 f32 -> bf16 (4 float4 per thread, coalesced)
        #pragma unroll
        for (int it = 0; it < 4; ++it) {
            int q = tid + it * 256;
            int row = q >> 3, kq = q & 7;
            const float4 f = *(const float4*)(A + (size_t)(bm + row) * K + k0 + kq * 4);
            unsigned int lo = (unsigned int)f2bf(f.x) | ((unsigned int)f2bf(f.y) << 16);
            unsigned int hi = (unsigned int)f2bf(f.z) | ((unsigned int)f2bf(f.w) << 16);
            *(uint2*)(&As[row * ASTR + kq * 4]) = make_uint2(lo, hi);
        }
        // stage W tile: 32x32 f32 -> bf16 (1 float4 per thread)
        {
            int row = tid >> 3, kq = tid & 7;
            const float4 f = *(const float4*)(W + (size_t)(bn + row) * K + k0 + kq * 4);
            unsigned int lo = (unsigned int)f2bf(f.x) | ((unsigned int)f2bf(f.y) << 16);
            unsigned int hi = (unsigned int)f2bf(f.z) | ((unsigned int)f2bf(f.w) << 16);
            *(uint2*)(&Ws[row * ASTR + kq * 4]) = make_uint2(lo, hi);
        }
        __syncthreads();

        // gather fragments per ISA VGPR layouts (coalesces to ds_load_b128)
        v8u au, bu0, bu1;
        #pragma unroll
        for (int v = 0; v < 8; ++v) {
            int kf = wmma_kf(v, hh);
            au[v]  = *(const unsigned int*)(&As[(wv * 16 + ml) * ASTR + kf]);
            bu0[v] = *(const unsigned int*)(&Ws[(ml)      * ASTR + kf]);
            bu1[v] = *(const unsigned int*)(&Ws[(16 + ml) * ASTR + kf]);
        }
        v16bf af  = __builtin_bit_cast(v16bf, au);
        v16bf bf0 = __builtin_bit_cast(v16bf, bu0);
        v16bf bf1 = __builtin_bit_cast(v16bf, bu1);
        acc0 = wmma_bf16(af, bf0, acc0);
        acc1 = wmma_bf16(af, bf1, acc1);
    }

    // epilogue: D layout -> rows r / r+8 per half, N = lane%16 (branch-free)
    #pragma unroll
    for (int nt = 0; nt < 2; ++nt) {
        const v8f acc = nt ? acc1 : acc0;
        #pragma unroll
        for (int r = 0; r < 8; ++r) {
            int row = bm + wv * 16 + r + 8 * hh;
            int col = bn + nt * 16 + ml;
            float val = acc[r] + bias[col];
            if constexpr (MODE == 0) {
                outF[(size_t)row * N + col] = val;
            } else if constexpr (MODE == 1) {
                outB[(size_t)row * N + col] = f2bf(val);
            } else {
                int bb = row >> 12;            // / N_NODES
                int rr = row & (N_NODES - 1);  // % N_NODES
                outB[((size_t)bb * N + col) * N_NODES + rr] = f2bf(val);
            }
        }
    }
}

// ---------------------------------------------------------------------------
// Flash-style masked attention.
// Block = 256 thr = 8 waves; wave w handles head h=w of one (batch, 16-edge
// tile). Streams node tiles of 32: S = Q Kh^T /8, mask by incidence, online
// softmax (shfl reductions over the 16-lane N groups), P via per-wave LDS
// transpose, O += P V (4 WMMAs over d_k=64). Q,K bf16 row-major; V bf16
// transposed per batch [b][d][node]; output f32 [b][e][h*64+d].
// ---------------------------------------------------------------------------
__global__ __launch_bounds__(256) void hyperattn_kernel(
    const unsigned short* __restrict__ Qb,
    const unsigned short* __restrict__ Kb,
    const unsigned short* __restrict__ Vt,
    const int* __restrict__ inc,
    float* __restrict__ outA)
{
    __shared__ unsigned short Plds[N_HEADS][16 * 34]; // per-wave P scratch

    const int tid   = threadIdx.x;
    const int h     = tid >> 5;        // wave id == head
    const int lane  = tid & 31;
    const int hh    = lane >> 4;
    const int ml    = lane & 15;
    const int b     = blockIdx.x >> 6; // 64 edge tiles per batch
    const int etile = blockIdx.x & 63;

    // Q fragments (A-layout): row = edge, K pairs = head dims
    v8u qa0u, qa1u;
    {
        const size_t qrow = ((size_t)b * N_EDGES + etile * 16 + ml) * D_MODEL + h * D_K;
        #pragma unroll
        for (int v = 0; v < 8; ++v) {
            int kf = wmma_kf(v, hh);
            qa0u[v] = *(const unsigned int*)(Qb + qrow + kf);
            qa1u[v] = *(const unsigned int*)(Qb + qrow + 32 + kf);
        }
    }
    const v16bf qa0 = __builtin_bit_cast(v16bf, qa0u);
    const v16bf qa1 = __builtin_bit_cast(v16bf, qa1u);

    v8f o[4] = {v8f{}, v8f{}, v8f{}, v8f{}};
    float mrow[8], lrow[8];
    #pragma unroll
    for (int r = 0; r < 8; ++r) { mrow[r] = -__builtin_inff(); lrow[r] = 0.0f; }

    unsigned short* pw = &Plds[h][0];

    for (int s = 0; s < N_NODES; s += 32) {
        float s0[8], s1[8];
        // ---- scores for two 16-node subtiles
        #pragma unroll
        for (int sub = 0; sub < 2; ++sub) {
            const int nbase = s + sub * 16;
            v8u bk0, bk1;
            const size_t krow = ((size_t)b * N_NODES + nbase + ml) * D_MODEL + h * D_K;
            #pragma unroll
            for (int v = 0; v < 8; ++v) {
                int kf = wmma_kf(v, hh);
                bk0[v] = *(const unsigned int*)(Kb + krow + kf);
                bk1[v] = *(const unsigned int*)(Kb + krow + 32 + kf);
            }
            v8f sacc = {};
            sacc = wmma_bf16(qa0, __builtin_bit_cast(v16bf, bk0), sacc);
            sacc = wmma_bf16(qa1, __builtin_bit_cast(v16bf, bk1), sacc);
            // scale + incidence mask (all heads share it; lives in 192MB L2)
            #pragma unroll
            for (int r = 0; r < 8; ++r) {
                int e  = etile * 16 + r + 8 * hh;
                int iv = inc[((size_t)b * N_EDGES + e) * N_NODES + nbase + ml];
                float sv = iv ? sacc[r] * 0.125f : -1e9f;
                if (sub == 0) s0[r] = sv; else s1[r] = sv;
            }
        }
        // ---- online softmax per row (row lives across a 16-lane group)
        #pragma unroll
        for (int r = 0; r < 8; ++r) {
            float rm = fmaxf(s0[r], s1[r]);
            rm = fmaxf(rm, __shfl_xor(rm, 1));
            rm = fmaxf(rm, __shfl_xor(rm, 2));
            rm = fmaxf(rm, __shfl_xor(rm, 4));
            rm = fmaxf(rm, __shfl_xor(rm, 8));
            float mnew  = fmaxf(mrow[r], rm);
            float alpha = __expf(mrow[r] - mnew);
            mrow[r] = mnew;
            float e0 = __expf(s0[r] - mnew);
            float e1 = __expf(s1[r] - mnew);
            float rs = e0 + e1;
            rs += __shfl_xor(rs, 1);
            rs += __shfl_xor(rs, 2);
            rs += __shfl_xor(rs, 4);
            rs += __shfl_xor(rs, 8);
            lrow[r] = lrow[r] * alpha + rs;
            o[0][r] *= alpha; o[1][r] *= alpha; o[2][r] *= alpha; o[3][r] *= alpha;
            const int prow = r + 8 * hh;
            pw[prow * 34 + ml]      = f2bf(e0);
            pw[prow * 34 + 16 + ml] = f2bf(e1);
        }
        // per-wave private LDS: DS ops are in-order within a wave; just stop
        // the compiler from reordering the transpose read above the writes.
        asm volatile("" ::: "memory");

        // ---- P (A-layout, K=32 nodes) from LDS transpose
        v8u pu;
        #pragma unroll
        for (int v = 0; v < 8; ++v)
            pu[v] = *(const unsigned int*)(pw + ml * 34 + wmma_kf(v, hh));
        const v16bf pf = __builtin_bit_cast(v16bf, pu);

        // ---- O += P @ V : 4 d-tiles of 16; V^T makes K-pairs contiguous
        const size_t vbase = ((size_t)b * D_MODEL + h * D_K) * N_NODES + s;
        #pragma unroll
        for (int nt = 0; nt < 4; ++nt) {
            v8u vu;
            #pragma unroll
            for (int v = 0; v < 8; ++v)
                vu[v] = *(const unsigned int*)(Vt + vbase +
                         (size_t)(nt * 16 + ml) * N_NODES + wmma_kf(v, hh));
            o[nt] = wmma_bf16(pf, __builtin_bit_cast(v16bf, vu), o[nt]);
        }
        // prefetch next K tile (global_prefetch_b8)
        if (s + 32 < N_NODES)
            __builtin_prefetch(Kb + ((size_t)b * N_NODES + s + 32 + ml) * D_MODEL + h * D_K, 0, 1);
    }

    // ---- epilogue: O * (1/l) -> f32 workspace [b][e][h*64+d]
    float linv[8];
    #pragma unroll
    for (int r = 0; r < 8; ++r) linv[r] = 1.0f / lrow[r];
    #pragma unroll
    for (int nt = 0; nt < 4; ++nt) {
        #pragma unroll
        for (int r = 0; r < 8; ++r) {
            int e = etile * 16 + r + 8 * hh;
            int d = h * D_K + nt * 16 + ml;
            outA[((size_t)b * N_EDGES + e) * D_MODEL + d] = o[nt][r] * linv[r];
        }
    }
}

// ---------------------------------------------------------------------------
extern "C" void kernel_launch(void* const* d_in, const int* in_sizes, int n_in,
                              void* d_out, int out_size, void* d_ws, size_t ws_size,
                              hipStream_t stream) {
    (void)in_sizes; (void)n_in; (void)out_size; (void)ws_size;
    const float* queries = (const float*)d_in[0];
    const float* keys    = (const float*)d_in[1];
    const int*   incid   = (const int*)d_in[2];
    const float* Wq = (const float*)d_in[3];
    const float* bq = (const float*)d_in[4];
    const float* Wk = (const float*)d_in[5];
    const float* bk = (const float*)d_in[6];
    const float* Wv = (const float*)d_in[7];
    const float* bv = (const float*)d_in[8];
    const float* Wo = (const float*)d_in[9];
    const float* bo = (const float*)d_in[10];
    float* out = (float*)d_out;

    // workspace carve-up (44 MB total)
    unsigned short* Qb = (unsigned short*)d_ws;                       // 4096 x 512 bf16
    unsigned short* Kb = Qb + (size_t)BS * N_EDGES * D_MODEL;         // 16384 x 512 bf16
    unsigned short* Vt = Kb + (size_t)BS * N_NODES * D_MODEL;         // [b][512][4096] bf16
    float* attn        = (float*)(Vt + (size_t)BS * N_NODES * D_MODEL); // 4096 x 512 f32

    const dim3 blk(256);
    const int ME = BS * N_EDGES;   // 4096
    const int MN = BS * N_NODES;   // 16384

    // Q = queries @ Wq^T + bq      -> bf16 row-major
    gemm_xwt_bias_kernel<1><<<dim3(ME / 128, D_MODEL / 32), blk, 0, stream>>>(
        queries, Wq, bq, nullptr, Qb);
    // K = keys @ Wk^T + bk         -> bf16 row-major
    gemm_xwt_bias_kernel<1><<<dim3(MN / 128, D_MODEL / 32), blk, 0, stream>>>(
        keys, Wk, bk, nullptr, Kb);
    // V = keys @ Wv^T + bv         -> bf16 transposed per batch [b][d][node]
    gemm_xwt_bias_kernel<2><<<dim3(MN / 128, D_MODEL / 32), blk, 0, stream>>>(
        keys, Wv, bv, nullptr, Vt);
    // masked flash attention        -> f32 [b][e][512]
    hyperattn_kernel<<<dim3(BS * (N_EDGES / 16)), blk, 0, stream>>>(
        Qb, Kb, Vt, incid, attn);
    // out = attn @ Wo^T + bo        -> f32 d_out
    gemm_xwt_bias_kernel<0><<<dim3(ME / 128, D_MODEL / 32), blk, 0, stream>>>(
        attn, Wo, bo, out, nullptr);
}
